// FrustumFeatureEncoder_32804960207436
// MI455X (gfx1250) — matrix operand order repaired and checked by reading, not compiled, not run on hardware
//
#include <hip/hip_runtime.h>
#include <math.h>

// ---------------------------------------------------------------------------
// CDNA5 (gfx1250) WMMA types
// ---------------------------------------------------------------------------
typedef __attribute__((ext_vector_type(16))) _Float16 v16h;
typedef __attribute__((ext_vector_type(8)))  float    v8f;

#define EPSBN 1e-5f

union V16H { v16h h; uint4 u4[2]; };

// A-matrix (16x32 f16) fragment load from a row-major [16][stride] f16 buffer.
// Per ISA 7.12.2: lanes 0-15 hold M=lane, K in {k0..k0+7, k0+16..k0+23};
// lanes 16-31 hold M=lane-16, K in {k0+8..k0+15, k0+24..k0+31}.
__device__ __forceinline__ v16h load_A16(const _Float16* row, int k0, int lane) {
  const _Float16* p0 = row + k0 + ((lane >> 4) << 3);
  V16H r;
  r.u4[0] = *(const uint4*)(p0);        // 8 halves (16 B)
  r.u4[1] = *(const uint4*)(p0 + 16);   // 8 halves (16 B)
  return r.h;
}

// B-matrix (32x16 f16) fragment load from row-major W[N][K] (N=out-ch, K=in-ch).
// Per ISA: lanes 0-15 hold N=lane, K in {k0..k0+7, k0+16..k0+23}; lanes 16-31
// hold N=lane-16, K in {k0+8..k0+15, k0+24..k0+31} (matches load_A16).
__device__ __forceinline__ v16h load_B16(const _Float16* __restrict__ W, int K,
                                         int n0, int k0, int lane) {
  const _Float16* p = W + (size_t)(n0 + (lane & 15)) * K + k0 + ((lane >> 4) << 3);
  V16H r;
  r.u4[0] = *(const uint4*)(p);
  r.u4[1] = *(const uint4*)(p + 16);
  return r.h;
}

__device__ __forceinline__ v8f wmma_f16(v16h a, v16h b, v8f c) {
  // (neg_a, A, neg_b, B, c_mod, C, reuse_a, reuse_b)
  return __builtin_amdgcn_wmma_f32_16x16x32_f16(false, a, false, b, (short)0, c,
                                                false, false);
}

// Splat a per-N bias into the accumulator: every element a lane owns shares
// the same column N, so the WMMA chain absorbs the bias for free.
__device__ __forceinline__ v8f bias_acc(float b) {
  v8f c;
#pragma unroll
  for (int r = 0; r < 8; ++r) c[r] = b;
  return c;
}

// Order-preserving float <-> u32 mapping so segment-max can use native u32
// atomics (global_atomic_max_u32) instead of a CAS loop.
__device__ __forceinline__ unsigned enc_f32(float x) {
  unsigned u = __float_as_uint(x);
  return (u & 0x80000000u) ? ~u : (u | 0x80000000u);
}
__device__ __forceinline__ float dec_f32(unsigned u) {
  unsigned b = (u & 0x80000000u) ? (u & 0x7FFFFFFFu) : ~u;
  return __uint_as_float(b);
}

// ---------------------------------------------------------------------------
// K0: init voxel accumulators. enc(-FLT_MAX) = 0x00800000 (reference floors
// empty voxels at finfo.min).
// ---------------------------------------------------------------------------
__global__ void __launch_bounds__(256)
ffe_init_kernel(unsigned* __restrict__ vmax, float* __restrict__ vsum,
                float* __restrict__ vcnt, int V) {
  int i = blockIdx.x * 256 + threadIdx.x;
  if (i < V * 256) vmax[i] = 0x00800000u;
  if (i < V * 4)   vsum[i] = 0.0f;
  if (i < V)       vcnt[i] = 0.0f;
}

// ---------------------------------------------------------------------------
// K1: fold BatchNorms into weights, convert to f16 (single block).
//   h0 = relu(feats @ W0f.T + c0)   W0f[o][i] = w0[o][i]*sp[i]*s0[o]  (K pad 32)
//   h1 = relu(h0 @ W1f.T + c1)      W1f[o][i] = w1[o][i]*s1[o]
//   h2 = h1 @ w2.T + b2             (w2 -> f16 copy)
// ---------------------------------------------------------------------------
__global__ void __launch_bounds__(256)
ffe_prep_kernel(const float* __restrict__ pn_g, const float* __restrict__ pn_b,
                const float* __restrict__ pn_m, const float* __restrict__ pn_v,
                const float* __restrict__ w0, const float* __restrict__ g0,
                const float* __restrict__ b0, const float* __restrict__ m0,
                const float* __restrict__ vv0,
                const float* __restrict__ w1, const float* __restrict__ g1,
                const float* __restrict__ b1, const float* __restrict__ m1,
                const float* __restrict__ vv1,
                const float* __restrict__ w2, const float* __restrict__ wc,
                _Float16* __restrict__ W0f, float* __restrict__ c0,
                _Float16* __restrict__ W1f, float* __restrict__ c1,
                _Float16* __restrict__ W2h, _Float16* __restrict__ Wch) {
  const int t = threadIdx.x;
  __shared__ float sp[8], tp[8];
  if (t < 8) {
    float s = pn_g[t] * rsqrtf(pn_v[t] + EPSBN);
    sp[t] = s;
    tp[t] = pn_b[t] - pn_m[t] * s;
  }
  __syncthreads();
  if (t < 64) {
    float s = g0[t] * rsqrtf(vv0[t] + EPSBN);
    float acc = 0.f;
#pragma unroll
    for (int i = 0; i < 8; ++i) {
      float w = w0[t * 8 + i];
      W0f[t * 32 + i] = (_Float16)(w * sp[i] * s);
      acc += tp[i] * w;
    }
#pragma unroll
    for (int i = 8; i < 32; ++i) W0f[t * 32 + i] = (_Float16)0.f;
    c0[t] = s * acc + b0[t] - m0[t] * s;
  }
  if (t < 128) {
    float s = g1[t] * rsqrtf(vv1[t] + EPSBN);
    for (int i = 0; i < 64; ++i)
      W1f[t * 64 + i] = (_Float16)(w1[t * 64 + i] * s);
    c1[t] = b1[t] - m1[t] * s;
  }
  for (int i = 0; i < 128; ++i)                      // 256 rows, t in [0,256)
    W2h[t * 128 + i] = (_Float16)w2[t * 128 + i];
  if (t < 64)
    for (int i = 0; i < 256; ++i)
      Wch[t * 256 + i] = (_Float16)wc[t * 256 + i];
}

// ---------------------------------------------------------------------------
// K2: per-voxel sums/counts (for cluster-center means).
// ---------------------------------------------------------------------------
__global__ void __launch_bounds__(256)
ffe_scatter_kernel(const float* __restrict__ pts, const int* __restrict__ imap,
                   int N, float* __restrict__ vsum, float* __restrict__ vcnt) {
  int i = blockIdx.x * 256 + threadIdx.x;
  if (i >= N) return;
  float4 p = ((const float4*)pts)[i];
  int v = imap[i];
  atomicAdd(&vsum[v * 4 + 0], p.x);
  atomicAdd(&vsum[v * 4 + 1], p.y);
  atomicAdd(&vsum[v * 4 + 2], p.z);
  atomicAdd(&vsum[v * 4 + 3], p.w);
  atomicAdd(&vcnt[v], 1.0f);
}

// ---------------------------------------------------------------------------
// K3: point MLP. One wave = 16 points. 8 waves/block = 128 points.
// GEMM0 (K=32 padded): 4 wmma; GEMM1 (K=64,N=128): 16 wmma;
// GEMM2 (K=128,N=256): 64 wmma; then atomic segment-max scatter.
// Biases ride in the WMMA C operand.
// ---------------------------------------------------------------------------
__global__ void __launch_bounds__(256)
ffe_point_kernel(const float* __restrict__ pts, const int* __restrict__ imap,
                 int N, const float* __restrict__ vsum,
                 const float* __restrict__ vcnt,
                 const _Float16* __restrict__ W0, const float* __restrict__ c0,
                 const _Float16* __restrict__ W1, const float* __restrict__ c1,
                 const _Float16* __restrict__ W2, const float* __restrict__ b2,
                 unsigned* __restrict__ vmax) {
  __shared__ alignas(16) _Float16 sF[8][16 * 32];
  __shared__ alignas(16) _Float16 sH0[8][16 * 64];
  __shared__ alignas(16) _Float16 sH1[8][16 * 128];
  __shared__ int sVox[8][16];

  const int wave = threadIdx.x >> 5;
  const int lane = threadIdx.x & 31;
  const int base = (blockIdx.x * 8 + wave) * 16;

  _Float16* fs = sF[wave];
  _Float16* h0 = sH0[wave];
  _Float16* h1 = sH1[wave];

  // ---- stage raw features (points, dist, cluster offset) into LDS as f16 ----
  if (lane < 16) {
    const int i = base + lane;
    float f[8] = {0.f, 0.f, 0.f, 0.f, 0.f, 0.f, 0.f, 0.f};
    int v = -1;
    if (i < N) {
      const float4 p = ((const float4*)pts)[i];
      v = imap[i];
      const float inv = 1.0f / vcnt[v];  // voxel referenced => cnt >= 1
      f[0] = p.x; f[1] = p.y; f[2] = p.z; f[3] = p.w;
      f[4] = sqrtf(p.x * p.x + p.y * p.y + p.z * p.z);
      f[5] = p.x - vsum[v * 4 + 0] * inv;
      f[6] = p.y - vsum[v * 4 + 1] * inv;
      f[7] = p.z - vsum[v * 4 + 2] * inv;
    }
    sVox[wave][lane] = v;
#pragma unroll
    for (int k = 0; k < 8; ++k)  fs[lane * 32 + k] = (_Float16)f[k];
#pragma unroll
    for (int k = 8; k < 32; ++k) fs[lane * 32 + k] = (_Float16)0.f;
  }
  __syncthreads();

  const int ln = lane & 15;          // this lane's output column within tile
  const int mb = (lane >> 4) << 3;   // M-row base of this lane's D elements

  // ---- GEMM0: feats[16x32] @ W0f.T -> h0[16][64], bias-in-C + relu ----
  {
    const v16h a = load_A16(fs + ln * 32, 0, lane);
#pragma unroll
    for (int nt = 0; nt < 4; ++nt) {
      const int n = nt * 16 + ln;
      v8f acc = bias_acc(c0[n]);
      acc = wmma_f16(a, load_B16(W0, 32, nt * 16, 0, lane), acc);
#pragma unroll
      for (int r = 0; r < 8; ++r)
        h0[(mb + r) * 64 + n] = (_Float16)fmaxf(acc[r], 0.f);
    }
  }
  __syncthreads();

  // ---- GEMM1: h0[16][64] @ W1f.T -> h1[16][128], bias-in-C + relu ----
  {
    const _Float16* arow = h0 + ln * 64;
    const v16h a0 = load_A16(arow, 0, lane);
    const v16h a1 = load_A16(arow, 32, lane);
#pragma unroll
    for (int nt = 0; nt < 8; ++nt) {
      const int n = nt * 16 + ln;
      v8f acc = bias_acc(c1[n]);
      acc = wmma_f16(a0, load_B16(W1, 64, nt * 16, 0, lane), acc);
      acc = wmma_f16(a1, load_B16(W1, 64, nt * 16, 32, lane), acc);
#pragma unroll
      for (int r = 0; r < 8; ++r)
        h1[(mb + r) * 128 + n] = (_Float16)fmaxf(acc[r], 0.f);
    }
  }
  __syncthreads();

  // ---- GEMM2: h1[16][128] @ w2.T + b2 -> atomic segment-max into vmax ----
  {
    const _Float16* arow = h1 + ln * 128;
    v16h a[4];
#pragma unroll
    for (int ks = 0; ks < 4; ++ks) a[ks] = load_A16(arow, ks * 32, lane);

    // hoist voxel ids / row offsets for the 8 M rows this lane produces
    long voff[8];
#pragma unroll
    for (int r = 0; r < 8; ++r) {
      const int v = sVox[wave][mb + r];
      voff[r] = (v >= 0) ? (long)v * 256 : -1;
    }

#pragma unroll
    for (int nt = 0; nt < 16; ++nt) {
      const int n = nt * 16 + ln;
      v8f acc = bias_acc(b2[n]);
#pragma unroll
      for (int ks = 0; ks < 4; ++ks)
        acc = wmma_f16(a[ks], load_B16(W2, 128, nt * 16, ks * 32, lane), acc);
#pragma unroll
      for (int r = 0; r < 8; ++r) {
        if (voff[r] >= 0)
          atomicMax(&vmax[voff[r] + n], enc_f32(acc[r]));
      }
    }
  }
}

// ---------------------------------------------------------------------------
// K4: voxel head: relu(vox_max[16x256] @ wc.T + bc) -> out[V][64].
// One wave = 16 voxels; 4 waves/block (32 KB LDS staging).
// ---------------------------------------------------------------------------
__global__ void __launch_bounds__(128)
ffe_voxel_kernel(const unsigned* __restrict__ vmax, int V,
                 const _Float16* __restrict__ Wc, const float* __restrict__ bc,
                 float* __restrict__ out) {
  __shared__ alignas(16) _Float16 sA[4][16 * 256];
  const int wave = threadIdx.x >> 5;
  const int lane = threadIdx.x & 31;
  const int base = (blockIdx.x * 4 + wave) * 16;
  _Float16* va = sA[wave];

  // stage decoded maxima into LDS (coalesced 32-lane channel sweeps)
  for (int r = 0; r < 16; ++r) {
    const int vid = base + r;
#pragma unroll
    for (int j = 0; j < 8; ++j) {
      const int ch = lane + (j << 5);
      float x = 0.f;
      if (vid < V) x = dec_f32(vmax[(size_t)vid * 256 + ch]);
      x = fmaxf(fminf(x, 60000.f), -60000.f);  // keep finite in f16
      va[r * 256 + ch] = (_Float16)x;
    }
  }
  __syncthreads();

  const int ln = lane & 15;
  const int mb = (lane >> 4) << 3;
  const _Float16* arow = va + ln * 256;
  v16h a[8];
#pragma unroll
  for (int ks = 0; ks < 8; ++ks) a[ks] = load_A16(arow, ks * 32, lane);
#pragma unroll
  for (int nt = 0; nt < 4; ++nt) {
    const int n = nt * 16 + ln;
    v8f acc = bias_acc(bc[n]);
#pragma unroll
    for (int ks = 0; ks < 8; ++ks)
      acc = wmma_f16(a[ks], load_B16(Wc, 256, nt * 16, ks * 32, lane), acc);
#pragma unroll
    for (int r = 0; r < 8; ++r) {
      const int vid = base + mb + r;
      if (vid < V) out[(size_t)vid * 64 + n] = fmaxf(acc[r], 0.f);
    }
  }
}

// ---------------------------------------------------------------------------
// Host-side launcher
// ---------------------------------------------------------------------------
extern "C" void kernel_launch(void* const* d_in, const int* in_sizes, int n_in,
                              void* d_out, int out_size, void* d_ws,
                              size_t ws_size, hipStream_t stream) {
  const float* points = (const float*)d_in[0];
  const int*   imap   = (const int*)d_in[1];
  // d_in[2] = num_voxels scalar (derived from out_size instead)
  const float* pn_g = (const float*)d_in[3];
  const float* pn_b = (const float*)d_in[4];
  const float* pn_m = (const float*)d_in[5];
  const float* pn_v = (const float*)d_in[6];
  const float* w0   = (const float*)d_in[7];
  const float* g0   = (const float*)d_in[8];
  const float* b0   = (const float*)d_in[9];
  const float* m0   = (const float*)d_in[10];
  const float* v0   = (const float*)d_in[11];
  const float* w1   = (const float*)d_in[12];
  const float* g1   = (const float*)d_in[13];
  const float* b1   = (const float*)d_in[14];
  const float* m1   = (const float*)d_in[15];
  const float* v1   = (const float*)d_in[16];
  const float* w2   = (const float*)d_in[17];
  const float* b2   = (const float*)d_in[18];
  const float* wc   = (const float*)d_in[19];
  const float* bc   = (const float*)d_in[20];
  float* out = (float*)d_out;

  const int N = in_sizes[0] / 4;   // points
  const int V = out_size / 64;     // voxels

  char* ws = (char*)d_ws;
  size_t off = 0;
  auto take = [&](size_t bytes) -> void* {
    void* p = (void*)(ws + off);
    off += (bytes + 255) & ~(size_t)255;
    return p;
  };
  unsigned* vmax = (unsigned*)take((size_t)V * 256 * 4);
  float*    vsum = (float*)take((size_t)V * 4 * 4);
  float*    vcnt = (float*)take((size_t)V * 4);
  _Float16* W0f  = (_Float16*)take(64 * 32 * 2);
  float*    c0   = (float*)take(64 * 4);
  _Float16* W1f  = (_Float16*)take(128 * 64 * 2);
  float*    c1   = (float*)take(128 * 4);
  _Float16* W2h  = (_Float16*)take(256 * 128 * 2);
  _Float16* Wch  = (_Float16*)take(64 * 256 * 2);
  (void)ws_size; (void)n_in;

  ffe_init_kernel<<<(V * 256 + 255) / 256, 256, 0, stream>>>(vmax, vsum, vcnt, V);
  ffe_prep_kernel<<<1, 256, 0, stream>>>(pn_g, pn_b, pn_m, pn_v,
                                         w0, g0, b0, m0, v0,
                                         w1, g1, b1, m1, v1,
                                         w2, wc, W0f, c0, W1f, c1, W2h, Wch);
  ffe_scatter_kernel<<<(N + 255) / 256, 256, 0, stream>>>(points, imap, N, vsum, vcnt);
  ffe_point_kernel<<<(N + 127) / 128, 256, 0, stream>>>(points, imap, N, vsum, vcnt,
                                                        W0f, c0, W1f, c1, W2h, b2,
                                                        vmax);
  ffe_voxel_kernel<<<(V + 63) / 64, 128, 0, stream>>>(vmax, V, Wch, bc, out);
}
